// Segmentor_62457414418643
// MI455X (gfx1250) — matrix operand order repaired and checked by reading, not compiled, not run on hardware
//
#include <hip/hip_runtime.h>
#include <hip/hip_bf16.h>

typedef __attribute__((ext_vector_type(16))) _Float16 v16h;
typedef __attribute__((ext_vector_type(8)))  _Float16 v8h;
typedef __attribute__((ext_vector_type(8)))  float    v8f;

#define H512 512
#define CIN 256
#define PD 32
#define SD 512
#define NPATCH 1024
#define PS 24
#define PATCH_ELEMS (PS*PS*PD)      // 18432
#define KCONV (9*CIN)               // 2304
#define KCCONV (KCONV/32)           // 72

__device__ __forceinline__ float gelu_tanh(float x) {
    float x3 = x * x * x;
    return 0.5f * x * (1.0f + tanhf(0.79788456080286535588f * (x + 0.044715f * x3)));
}
__device__ __forceinline__ float sigmoidf(float x) { return 1.0f / (1.0f + expf(-x)); }
__device__ __forceinline__ int iclamp(int v, int lo, int hi) { return v < lo ? lo : (v > hi ? hi : v); }

// ---------------------------------------------------------------- pack f32 [K][N] weight into WMMA B-fragment order
// Bp element (v16h units): (ntile*KC + kc)*32 + lane ; lane's fragment: h[2v+u] = B[kc*32 + (v>>2)*16 + (lane>>4)*8 + 2(v&3) + u][ntile*16 + (lane&15)]
__global__ void pack_b_k(const float* __restrict__ B, _Float16* __restrict__ Bp, int K, int N) {
    int idx = blockIdx.x * 256 + threadIdx.x;          // over (N/16)*KC*32
    int KC = K >> 5;
    int total = (N >> 4) * KC * 32;
    if (idx >= total) return;
    int lane = idx & 31;
    int kc = (idx >> 5) % KC;
    int ntile = (idx >> 5) / KC;
    int half = lane >> 4, n = ntile * 16 + (lane & 15);
    _Float16* o = Bp + (size_t)idx * 16;
#pragma unroll
    for (int v = 0; v < 8; ++v) {
        int kb = kc * 32 + (v >> 2) * 16 + half * 8 + 2 * (v & 3);
        o[2 * v]     = (_Float16)B[(size_t)kb * N + n];
        o[2 * v + 1] = (_Float16)B[(size_t)(kb + 1) * N + n];
    }
}

// ---------------------------------------------------------------- implicit-GEMM 3x3 conv 256->32 via WMMA
// One wave = 16 pixels x 32 out-channels. Tap-major loop: boundary predicate constant over the 256-ch inner loop.
__global__ void conv_gemm_k(const float* __restrict__ feat, const _Float16* __restrict__ Bp,
                            const float* __restrict__ bias, float* __restrict__ out) {
    int wave = threadIdx.x >> 5;
    int tm = blockIdx.x * 4 + wave;                    // 16384 M-tiles
    int lane = threadIdx.x & 31;
    int half = lane >> 4, r = lane & 15;
    int m = tm * 16 + r;
    int py = m >> 9, px = m & 511;                     // py uniform across the tile (tiles align to rows)
    v8f acc[2] = {{}, {}};
    const v16h* BpV = (const v16h*)Bp;
    for (int tap = 0; tap < 9; ++tap) {
        int ky = tap / 3, kx = tap - 3 * (tap / 3);
        int y = py + ky - 1, x = px + kx - 1;
        bool valid = (y >= 0) && (y < H512) && (x >= 0) && (x < H512);
        int yc = iclamp(y, 0, H512 - 1), xc = iclamp(x, 0, H512 - 1);
        const float* basef = feat + (((size_t)yc << 9) + xc) * CIN;
        __builtin_prefetch(basef + CIN, 0, 3);
#pragma unroll
        for (int c8 = 0; c8 < 8; ++c8) {
            int k0 = c8 * 32;
            float4 f0 = *(const float4*)(basef + k0 + half * 8);
            float4 f1 = *(const float4*)(basef + k0 + half * 8 + 4);
            float4 f2 = *(const float4*)(basef + k0 + 16 + half * 8);
            float4 f3 = *(const float4*)(basef + k0 + 16 + half * 8 + 4);
            v16h a;
            a[0]=(_Float16)f0.x; a[1]=(_Float16)f0.y; a[2]=(_Float16)f0.z; a[3]=(_Float16)f0.w;
            a[4]=(_Float16)f1.x; a[5]=(_Float16)f1.y; a[6]=(_Float16)f1.z; a[7]=(_Float16)f1.w;
            a[8]=(_Float16)f2.x; a[9]=(_Float16)f2.y; a[10]=(_Float16)f2.z; a[11]=(_Float16)f2.w;
            a[12]=(_Float16)f3.x; a[13]=(_Float16)f3.y; a[14]=(_Float16)f3.z; a[15]=(_Float16)f3.w;
            v16h z = {};
            a = valid ? a : z;                          // branchless zero-fill for OOB taps
            int kc = tap * 8 + c8;
            const v16h* bp = BpV + (size_t)kc * 32 + lane;
            v16h b0 = bp[0];
            v16h b1 = bp[(size_t)KCCONV * 32];
            acc[0] = __builtin_amdgcn_wmma_f32_16x16x32_f16(false, a, false, b0, (short)0, acc[0], false, false);
            acc[1] = __builtin_amdgcn_wmma_f32_16x16x32_f16(false, a, false, b1, (short)0, acc[1], false, false);
        }
    }
#pragma unroll
    for (int t = 0; t < 2; ++t) {
        int on = t * 16 + r;
        float bs = bias[on];
#pragma unroll
        for (int rr = 0; rr < 8; ++rr) {
            int om = tm * 16 + half * 8 + rr;
            out[(size_t)om * PD + on] = acc[t][rr] + bs;
        }
    }
}

// ---------------------------------------------------------------- dense WMMA GEMM: one wave = 16x64 tile (4 accumulators)
// A: f16 row-major [M][K]; Bp: packed fragments; C = act(A@B + bias)
__global__ void gemm_wmma_k(const _Float16* __restrict__ A, const _Float16* __restrict__ Bp,
                            const float* __restrict__ bias, float* __restrict__ Cf32,
                            _Float16* __restrict__ Cf16, int M, int N, int K, int act) {
    int KC = K >> 5;
    int tiles_g = N >> 6;                               // groups of 4 n-tiles
    int wave = threadIdx.x >> 5;
    int tile = blockIdx.x * 4 + wave;
    int tm = tile / tiles_g, tg = tile % tiles_g;
    int lane = threadIdx.x & 31;
    int half = lane >> 4, r = lane & 15;
    const _Float16* Arow = A + (size_t)(tm * 16 + r) * K;
    const v16h* BpV = (const v16h*)Bp;
    v8f acc[4] = {{}, {}, {}, {}};
    for (int kc = 0; kc < KC; ++kc) {
        int k0 = kc * 32;
        v8h alo = *(const v8h*)(Arow + k0 + half * 8);
        v8h ahi = *(const v8h*)(Arow + k0 + 16 + half * 8);
        v16h a = __builtin_shufflevector(alo, ahi, 0,1,2,3,4,5,6,7,8,9,10,11,12,13,14,15);
        __builtin_prefetch(Arow + k0 + 64, 0, 3);
        const v16h* bp = BpV + ((size_t)(tg * 4) * KC + kc) * 32 + lane;
#pragma unroll
        for (int t = 0; t < 4; ++t) {
            v16h b = bp[(size_t)t * KC * 32];
            acc[t] = __builtin_amdgcn_wmma_f32_16x16x32_f16(false, a, false, b, (short)0, acc[t], false, false);
        }
    }
#pragma unroll
    for (int t = 0; t < 4; ++t) {
        int on = (tg * 4 + t) * 16 + r;
        float bs = bias ? bias[on] : 0.0f;
#pragma unroll
        for (int rr = 0; rr < 8; ++rr) {
            int om = tm * 16 + half * 8 + rr;
            float v = acc[t][rr] + bs;
            if (act == 1) v = gelu_tanh(v);
            if (Cf32) Cf32[(size_t)om * N + on] = v;
            if (Cf16) Cf16[(size_t)om * N + on] = (_Float16)v;
        }
    }
}

// ---------------------------------------------------------------- gather 24x24x32 patches + per-channel mean
__global__ void gather_mean_k(const float* __restrict__ featpd, const float* __restrict__ loc,
                              float* __restrict__ patches, float* __restrict__ meanb) {
    __shared__ float ssum[PD];
    int p = blockIdx.x, t = threadIdx.x;
    if (t < PD) ssum[t] = 0.0f;
    __syncthreads();
    float ly = loc[2 * p] * 0.25f, lx = loc[2 * p + 1] * 0.25f;
    int y0 = (int)ly - PS / 2, x0 = (int)lx - PS / 2;
    float accum = 0.0f;
    float* pp = patches + (size_t)p * PATCH_ELEMS;
    for (int e = t; e < PATCH_ELEMS; e += 256) {
        int ch = e & 31, pix = e >> 5;
        int iy = pix / PS, ix = pix - PS * iy;
        int gy = y0 + iy, gx = x0 + ix;
        float v = 0.0f;
        if (gy >= 0 && gy < H512 && gx >= 0 && gx < H512)
            v = featpd[((size_t)gy * H512 + gx) * PD + ch];
        pp[e] = v;
        accum += v;
    }
    atomicAdd(&ssum[t & 31], accum);
    __syncthreads();
    if (t < PD) meanb[p * PD + t] = ssum[t] * (1.0f / (PS * PS));
}

// ---------------------------------------------------------------- squeeze-excitation + in-place scale
__global__ void se_scale_k(float* __restrict__ patches, const float* __restrict__ meanb,
                           const float* __restrict__ w1, const float* __restrict__ b1,
                           const float* __restrict__ w2, const float* __restrict__ b2) {
    __shared__ float y1[2];
    __shared__ float sc[PD];
    int p = blockIdx.x, t = threadIdx.x;
    if (t < 2) {
        float a = b1[t];
        for (int c = 0; c < PD; ++c) a += meanb[p * PD + c] * w1[c * 2 + t];
        y1[t] = a > 0.0f ? a : 0.0f;
    }
    __syncthreads();
    if (t < PD) sc[t] = sigmoidf(b2[t] + y1[0] * w2[t] + y1[1] * w2[PD + t]);
    __syncthreads();
    float s = sc[t & 31];
    float* pp = patches + (size_t)p * PATCH_ELEMS;
    for (int e = t; e < PATCH_ELEMS; e += 256) pp[e] *= s;
}

// ---------------------------------------------------------------- center crop 12x12 -> bilinear 8x8, pack f16 rows
__global__ void sig_resize_k(const float* __restrict__ patches, _Float16* __restrict__ s_in) {
    int p = blockIdx.x, t = threadIdx.x;
    const float* pp = patches + (size_t)p * PATCH_ELEMS;
    for (int e = t; e < 2048; e += 256) {
        int ch = e & 31, pix = e >> 5, oy = pix >> 3, ox = pix & 7;
        float sy = (oy + 0.5f) * 1.5f - 0.5f;
        float sx = (ox + 0.5f) * 1.5f - 0.5f;
        int fy = (int)floorf(sy), fx = (int)floorf(sx);
        float wy = sy - fy, wx = sx - fx;
        int iy0 = 6 + fy, iy1 = iy0 + 1, ix0 = 6 + fx, ix1 = ix0 + 1;
        float v00 = pp[(iy0 * PS + ix0) * PD + ch], v01 = pp[(iy0 * PS + ix1) * PD + ch];
        float v10 = pp[(iy1 * PS + ix0) * PD + ch], v11 = pp[(iy1 * PS + ix1) * PD + ch];
        float v = v00 * (1 - wy) * (1 - wx) + v01 * (1 - wy) * wx
                + v10 * wy * (1 - wx)       + v11 * wy * wx;
        s_in[(size_t)p * 2048 + e] = (_Float16)v;
    }
}

// ---------------------------------------------------------------- row layernorm (population var), dual f32/f16 out
__global__ void layernorm_k(const float* __restrict__ sp, const float* __restrict__ ln_s,
                            const float* __restrict__ ln_b, float* __restrict__ out_sig,
                            _Float16* __restrict__ sig16) {
    __shared__ float red[256], red2[256];
    int p = blockIdx.x, t = threadIdx.x;
    float a0 = sp[(size_t)p * SD + t], a1 = sp[(size_t)p * SD + 256 + t];
    red[t] = a0 + a1; red2[t] = a0 * a0 + a1 * a1;
    __syncthreads();
    for (int s = 128; s > 0; s >>= 1) {
        if (t < s) { red[t] += red[t + s]; red2[t] += red2[t + s]; }
        __syncthreads();
    }
    float m = red[0] * (1.0f / SD);
    float var = red2[0] * (1.0f / SD) - m * m;
    float inv = rsqrtf(var + 1e-6f);
    float o0 = (a0 - m) * inv * ln_s[t] + ln_b[t];
    float o1 = (a1 - m) * inv * ln_s[256 + t] + ln_b[256 + t];
    out_sig[(size_t)p * SD + t] = o0;         out_sig[(size_t)p * SD + 256 + t] = o1;
    sig16[(size_t)p * SD + t] = (_Float16)o0; sig16[(size_t)p * SD + 256 + t] = (_Float16)o1;
}

// ---------------------------------------------------------------- pe 16x16x4 -> bilinear 24x24, @enc_w, add + relu
__global__ void enc_add_relu_k(float* __restrict__ patches, const float* __restrict__ pe,
                               const float* __restrict__ enc_w) {
    int p = blockIdx.x, t = threadIdx.x;
    float* pp = patches + (size_t)p * PATCH_ELEMS;
    const float* pep = pe + (size_t)p * 1024;
    for (int e = t; e < PATCH_ELEMS; e += 256) {
        int ch = e & 31, pix = e >> 5;
        int iy = pix / PS, ix = pix - PS * iy;
        float sy = (iy + 0.5f) * (16.0f / 24.0f) - 0.5f;
        float sx = (ix + 0.5f) * (16.0f / 24.0f) - 0.5f;
        int fy = (int)floorf(sy), fx = (int)floorf(sx);
        float wy = sy - fy, wx = sx - fx;
        int y0 = iclamp(fy, 0, 15), y1 = iclamp(fy + 1, 0, 15);
        int x0 = iclamp(fx, 0, 15), x1 = iclamp(fx + 1, 0, 15);
        float acc = 0.0f;
#pragma unroll
        for (int j = 0; j < 4; ++j) {
            float v00 = pep[(y0 * 16 + x0) * 4 + j], v01 = pep[(y0 * 16 + x1) * 4 + j];
            float v10 = pep[(y1 * 16 + x0) * 4 + j], v11 = pep[(y1 * 16 + x1) * 4 + j];
            float v = v00 * (1 - wy) * (1 - wx) + v01 * (1 - wy) * wx
                    + v10 * wy * (1 - wx)       + v11 * wy * wx;
            acc += v * enc_w[j * PD + ch];
        }
        float rlu = pp[e] + acc;
        pp[e] = rlu > 0.0f ? rlu : 0.0f;
    }
}

// ---------------------------------------------------------------- conv_transpose 3x3 s2 (24->48 in LDS), bilinear 48->96
__global__ void convt_resize_k(const float* __restrict__ patches, const float* __restrict__ ct_w,
                               const float* __restrict__ ct_b, float* __restrict__ out) {
    __shared__ float up[48 * 48];
    int p = blockIdx.x, t = threadIdx.x;
    const float* pp = patches + (size_t)p * PATCH_ELEMS;
    float bias = ct_b[0];
    for (int o = t; o < 48 * 48; o += 256) {
        int oy = o / 48, ox = o - 48 * (o / 48);
        float acc = bias;
#pragma unroll
        for (int ky = 0; ky < 3; ++ky) {
            int ty = oy + 1 - ky;
            if (ty < 0 || (ty & 1)) continue;
            int iy = ty >> 1; if (iy >= PS) continue;
#pragma unroll
            for (int kx = 0; kx < 3; ++kx) {
                int tx = ox + 1 - kx;
                if (tx < 0 || (tx & 1)) continue;
                int ix = tx >> 1; if (ix >= PS) continue;
                const float* src = pp + (iy * PS + ix) * PD;
                const float* w = ct_w + (ky * 3 + kx) * PD;
#pragma unroll
                for (int c = 0; c < PD; ++c) acc += src[c] * w[c];
            }
        }
        up[o] = acc;
    }
    __syncthreads();
    float* op = out + (size_t)p * 96 * 96;
    for (int o = t; o < 96 * 96; o += 256) {
        int oy = o / 96, ox = o - 96 * (o / 96);
        float sy = oy * 0.5f - 0.25f, sx = ox * 0.5f - 0.25f;
        int fy = (int)floorf(sy), fx = (int)floorf(sx);
        float wy = sy - fy, wx = sx - fx;
        int y0 = iclamp(fy, 0, 47), y1 = iclamp(fy + 1, 0, 47);
        int x0 = iclamp(fx, 0, 47), x1 = iclamp(fx + 1, 0, 47);
        op[o] = up[y0 * 48 + x0] * (1 - wy) * (1 - wx) + up[y0 * 48 + x1] * (1 - wy) * wx
              + up[y1 * 48 + x0] * wy * (1 - wx)       + up[y1 * 48 + x1] * wy * wx;
    }
}

// ---------------------------------------------------------------- metadata outputs
__global__ void meta_k(const float* __restrict__ loc, float* __restrict__ out) {
    int i = blockIdx.x * 256 + threadIdx.x;
    if (i >= NPATCH) return;
    float ly = loc[2 * i] * 0.25f, lx = loc[2 * i + 1] * 0.25f;
    bool valid = lx >= 0.0f;
    int y0 = ((int)ly - PS / 2) * 4, x0 = ((int)lx - PS / 2) * 4;
    out[i]        = valid ? 1.0f : 0.0f;
    out[1024 + i] = valid ? (float)x0 : -1.0f;
    out[2048 + i] = valid ? (float)y0 : -1.0f;
    out[3072 + i] = 0.0f;
}

// ================================================================ host launch
extern "C" void kernel_launch(void* const* d_in, const int* in_sizes, int n_in,
                              void* d_out, int out_size, void* d_ws, size_t ws_size,
                              hipStream_t stream) {
    const float* feature = (const float*)d_in[0];
    const float* locations = (const float*)d_in[1];
    const float* conv_w = (const float*)d_in[2];
    const float* conv_b = (const float*)d_in[3];
    const float* ca_w1 = (const float*)d_in[4];
    const float* ca_b1 = (const float*)d_in[5];
    const float* ca_w2 = (const float*)d_in[6];
    const float* ca_b2 = (const float*)d_in[7];
    const float* sig_w = (const float*)d_in[8];
    const float* sig_b = (const float*)d_in[9];
    const float* ln_s = (const float*)d_in[10];
    const float* ln_b = (const float*)d_in[11];
    const float* pe_w1 = (const float*)d_in[12];
    const float* pe_b1 = (const float*)d_in[13];
    const float* pe_w2 = (const float*)d_in[14];
    const float* pe_b2 = (const float*)d_in[15];
    const float* pe_w3 = (const float*)d_in[16];
    const float* pe_b3 = (const float*)d_in[17];
    const float* enc_w = (const float*)d_in[18];
    const float* ct_w = (const float*)d_in[19];
    const float* ct_b = (const float*)d_in[20];

    char* ws = (char*)d_ws;
    size_t off = 0;
    auto take = [&](size_t bytes) { char* p = ws + off; off += (bytes + 255) & ~(size_t)255; return p; };
    _Float16* bp_conv = (_Float16*)take((size_t)KCONV * PD * 2);
    _Float16* bp_sig  = (_Float16*)take((size_t)2048 * SD * 2);
    _Float16* bp_pe1  = (_Float16*)take((size_t)SD * SD * 2);
    _Float16* bp_pe2  = (_Float16*)take((size_t)SD * SD * 2);
    _Float16* bp_pe3  = (_Float16*)take((size_t)SD * 1024 * 2);
    float*    feat_out = (float*)take((size_t)H512 * H512 * PD * 4);
    float*    patches  = (float*)take((size_t)NPATCH * PATCH_ELEMS * 4);
    float*    meanb    = (float*)take((size_t)NPATCH * PD * 4);
    _Float16* s_in     = (_Float16*)take((size_t)NPATCH * 2048 * 2);
    float*    s_pre    = (float*)take((size_t)NPATCH * SD * 4);
    _Float16* sig16    = (_Float16*)take((size_t)NPATCH * SD * 2);
    _Float16* x1_16    = (_Float16*)take((size_t)NPATCH * SD * 2);
    _Float16* x2_16    = (_Float16*)take((size_t)NPATCH * SD * 2);
    float*    pe_out   = (float*)take((size_t)NPATCH * 1024 * 4);
    (void)ws_size; (void)n_in; (void)in_sizes; (void)out_size;

    float* out = (float*)d_out;
    float* out_seg = out + 4096;                             // [1024][96][96]
    float* out_sig = out + 4096 + (size_t)NPATCH * 96 * 96;  // [1024][512]

    // 1. pack all weights into WMMA B-fragment layouts
    pack_b_k<<<(2  * KCCONV * 32 + 255) / 256, 256, 0, stream>>>(conv_w, bp_conv, KCONV, PD);
    pack_b_k<<<(32 * 64 * 32 + 255) / 256, 256, 0, stream>>>(sig_w, bp_sig, 2048, SD);
    pack_b_k<<<(32 * 16 * 32 + 255) / 256, 256, 0, stream>>>(pe_w1, bp_pe1, SD, SD);
    pack_b_k<<<(32 * 16 * 32 + 255) / 256, 256, 0, stream>>>(pe_w2, bp_pe2, SD, SD);
    pack_b_k<<<(64 * 16 * 32 + 255) / 256, 256, 0, stream>>>(pe_w3, bp_pe3, SD, 1024);

    // 2. implicit-GEMM conv via WMMA: 16384 tiles, 4 waves/block
    conv_gemm_k<<<(H512 * H512 / 16) / 4, 128, 0, stream>>>(feature, bp_conv, conv_b, feat_out);

    // 3. patch gather + channel means
    gather_mean_k<<<NPATCH, 256, 0, stream>>>(feat_out, locations, patches, meanb);

    // 4. squeeze-excitation scale (in place)
    se_scale_k<<<NPATCH, 256, 0, stream>>>(patches, meanb, ca_w1, ca_b1, ca_w2, ca_b2);

    // 5. signature input: crop + bilinear 8x8, f16
    sig_resize_k<<<NPATCH, 256, 0, stream>>>(patches, s_in);

    // 6. signature GEMM [1024x2048]@[2048x512] via WMMA (16x64 per wave)
    gemm_wmma_k<<<(64 * 8) / 4, 128, 0, stream>>>(s_in, bp_sig, sig_b, s_pre, nullptr,
                                                  NPATCH, SD, 2048, 0);
    // 7. layernorm -> d_out sig + f16 feed
    layernorm_k<<<NPATCH, 256, 0, stream>>>(s_pre, ln_s, ln_b, out_sig, sig16);

    // 8. PE MLP via WMMA (gelu, gelu, linear)
    gemm_wmma_k<<<(64 * 8) / 4, 128, 0, stream>>>(sig16, bp_pe1, pe_b1, nullptr, x1_16,
                                                  NPATCH, SD, SD, 1);
    gemm_wmma_k<<<(64 * 8) / 4, 128, 0, stream>>>(x1_16, bp_pe2, pe_b2, nullptr, x2_16,
                                                  NPATCH, SD, SD, 1);
    gemm_wmma_k<<<(64 * 16) / 4, 128, 0, stream>>>(x2_16, bp_pe3, pe_b3, pe_out, nullptr,
                                                   NPATCH, 1024, SD, 0);
    // 9. pe upsample + enc_w + relu (in place on patches)
    enc_add_relu_k<<<NPATCH, 256, 0, stream>>>(patches, pe_out, enc_w);

    // 10. transposed conv + bilinear upsample to 96x96 -> segmentations
    convt_resize_k<<<NPATCH, 256, 0, stream>>>(patches, ct_w, ct_b, out_seg);

    // 11. metadata (is_valid, x0, y0, z0)
    meta_k<<<4, 256, 0, stream>>>(locations, out);
}
